// Block_59330678227087
// MI455X (gfx1250) — matrix operand order repaired and checked by reading, not compiled, not run on hardware
//
#include <hip/hip_runtime.h>
#include <hip/hip_bf16.h>

// ---------------------------------------------------------------------------
// ViT block for MI455X (gfx1250): bf16 WMMA (f32 accum) everywhere.
// ---------------------------------------------------------------------------

typedef __bf16 bf16_t;
typedef __attribute__((ext_vector_type(16))) __bf16 v16bf;
typedef __attribute__((ext_vector_type(8)))  __bf16 v8bf;
typedef __attribute__((ext_vector_type(8)))  float  v8f;

#define DIM    768
#define HEADS  12
#define HD     64
#define HIDDEN 3072
#define BATCH  8
#define SEQ    1024
#define TOKENS (BATCH * SEQ)   // 8192

union afrag_u { v16bf v; v8bf h[2]; };

// ---- WMMA fragment helpers (wave32, 16x16x32 bf16) ------------------------
// A-frag (16x32, row m = lane&15): VGPRs 0-3 -> k = half*16 + 0..7,
// VGPRs 4-7 -> k = half*16 + 8..15  (half = lane>>4). Two 16B loads.
static __device__ inline v16bf load_a_frag(const bf16_t* A, int lda, int kbase, int lane) {
  const int m    = lane & 15;
  const int hsel = lane >> 4;
  const bf16_t* p = A + (size_t)m * lda + kbase + hsel * 16;
  afrag_u r;
  r.h[0] = *(const v8bf*)p;        // k = kbase + hsel*16 .. +7
  r.h[1] = *(const v8bf*)(p + 8);  // k = kbase + hsel*16 + 8 .. +15
  return r.v;
}

// B-frag (32x16, col n = lane&15): lanes 0-15 hold K=0..15, lanes 16-31 hold
// K=16..31, contiguous per lane. Bt is stored [N][K] so this is one 32B load.
static __device__ inline v16bf load_b_frag(const bf16_t* Bt, int ldb, int kbase, int lane) {
  const int n    = lane & 15;
  const int hsel = lane >> 4;
  return *(const v16bf*)(Bt + (size_t)n * ldb + kbase + hsel * 16);
}

static __device__ inline v8f wmma_bf16(v16bf a, v16bf b, v8f c) {
  // (neg_a, A, neg_b, B, c_mod, C, reuse_a, reuse_b)
  return __builtin_amdgcn_wmma_f32_16x16x32_bf16(false, a, false, b, (short)0, c,
                                                 false, false);
}

// ---- weight transpose + f32->bf16 -----------------------------------------
__global__ void transpose_cast_kernel(const float* __restrict__ w,
                                      bf16_t* __restrict__ wT, int K, int N) {
  size_t idx = (size_t)blockIdx.x * blockDim.x + threadIdx.x;
  if (idx >= (size_t)K * N) return;
  int n = (int)(idx / K);
  int k = (int)(idx % K);
  wT[idx] = (bf16_t)w[(size_t)k * N + n];
}

// ---- LayerNorm over C=768, one token per block, output bf16 ---------------
__global__ __launch_bounds__(256) void ln_kernel(const float* __restrict__ x,
                                                 const float* __restrict__ g,
                                                 const float* __restrict__ bta,
                                                 bf16_t* __restrict__ out) {
  const int row = blockIdx.x;
  const int tid = threadIdx.x;
  const float* xr = x + (size_t)row * DIM;
  float v0 = xr[tid], v1 = xr[tid + 256], v2 = xr[tid + 512];
  __shared__ float red[256];
  red[tid] = v0 + v1 + v2;
  __syncthreads();
#pragma unroll
  for (int s = 128; s > 0; s >>= 1) {
    if (tid < s) red[tid] += red[tid + s];
    __syncthreads();
  }
  float mu = red[0] * (1.0f / DIM);
  __syncthreads();
  float d0 = v0 - mu, d1 = v1 - mu, d2 = v2 - mu;
  red[tid] = d0 * d0 + d1 * d1 + d2 * d2;
  __syncthreads();
#pragma unroll
  for (int s = 128; s > 0; s >>= 1) {
    if (tid < s) red[tid] += red[tid + s];
    __syncthreads();
  }
  float rstd = rsqrtf(red[0] * (1.0f / DIM) + 1e-3f);
  bf16_t* orow = out + (size_t)row * DIM;
  orow[tid]       = (bf16_t)(d0 * rstd * g[tid]       + bta[tid]);
  orow[tid + 256] = (bf16_t)(d1 * rstd * g[tid + 256] + bta[tid + 256]);
  orow[tid + 512] = (bf16_t)(d2 * rstd * g[tid + 512] + bta[tid + 512]);
}

// ---- Tiled bf16 GEMM: C[M,N] = A[M,K] * Bt[N,K]^T + bias, fused epilogue --
// MODE 0: store bf16 (acc+bias)
// MODE 1: store f32  resid + gamma * (acc+bias)
// MODE 2: store bf16 gelu_exact(acc+bias)
// Block = 256 threads (8 waves); block tile 128x128; wave tile 32x64
// (2x4 accum tiles of 16x16). K step 32. All dims are multiples of tile sizes.
template <int MODE>
__global__ __launch_bounds__(256) void gemm_bf16_kernel(
    const bf16_t* __restrict__ A, const bf16_t* __restrict__ Bt,
    const float* __restrict__ bias, const float* __restrict__ resid,
    const float* __restrict__ gamma, bf16_t* __restrict__ outB,
    float* __restrict__ outF, int M, int N, int K) {
  const int lane = threadIdx.x & 31;
  const int wave = threadIdx.x >> 5;
  const int wm = wave >> 1, wn = wave & 1;
  const int rowBase = blockIdx.y * 128 + wm * 32;
  const int colBase = blockIdx.x * 128 + wn * 64;
  const int hsel = lane >> 4, l16 = lane & 15;

  v8f zf = {};
  v8f acc[2][4];
#pragma unroll
  for (int i = 0; i < 2; ++i)
#pragma unroll
    for (int j = 0; j < 4; ++j) acc[i][j] = zf;

  const bf16_t* Arow0 = A + (size_t)rowBase * K;
  const bf16_t* Arow1 = A + (size_t)(rowBase + 16) * K;

  for (int kk = 0; kk < K; kk += 32) {
    v16bf a0 = load_a_frag(Arow0, K, kk, lane);
    v16bf a1 = load_a_frag(Arow1, K, kk, lane);
#pragma unroll
    for (int j = 0; j < 4; ++j) {
      v16bf b = load_b_frag(Bt + (size_t)(colBase + j * 16) * K, K, kk, lane);
      acc[0][j] = wmma_bf16(a0, b, acc[0][j]);
      acc[1][j] = wmma_bf16(a1, b, acc[1][j]);
    }
  }

#pragma unroll
  for (int i = 0; i < 2; ++i)
#pragma unroll
    for (int j = 0; j < 4; ++j) {
      const int col = colBase + j * 16 + l16;
      const float bv = bias[col];
#pragma unroll
      for (int r = 0; r < 8; ++r) {
        const int row = rowBase + i * 16 + r + 8 * hsel;  // C-layout row
        const size_t o = (size_t)row * N + col;
        float val = acc[i][j][r] + bv;
        if (MODE == 0) {
          outB[o] = (bf16_t)val;
        } else if (MODE == 1) {
          outF[o] = resid[o] + gamma[col] * val;
        } else {  // exact GELU: 0.5*x*(1+erf(x/sqrt(2)))
          float gl = 0.5f * val * (1.0f + erff(val * 0.70710678118654752f));
          outB[o] = (bf16_t)gl;
        }
      }
    }
}

// ---- qkv [8192][2304] -> q [B,H,N,hd], k [B,H,N,hd], vT [B,H,hd,N] --------
__global__ void repack_qkv_kernel(const bf16_t* __restrict__ qkv,
                                  bf16_t* __restrict__ q,
                                  bf16_t* __restrict__ k,
                                  bf16_t* __restrict__ vT) {
  size_t idx = (size_t)blockIdx.x * blockDim.x + threadIdx.x;
  if (idx >= (size_t)TOKENS * DIM) return;
  int tok = (int)(idx / DIM);
  int c   = (int)(idx % DIM);
  int h = c / HD, d = c % HD;
  int b = tok / SEQ, n = tok % SEQ;
  size_t src = (size_t)tok * (3 * DIM);
  size_t bh  = (size_t)b * HEADS + h;
  size_t qkd = (bh * SEQ + n) * HD + d;
  q[qkd] = qkv[src + c];
  k[qkd] = qkv[src + DIM + c];
  vT[(bh * HD + d) * SEQ + n] = qkv[src + 2 * DIM + c];
}

// ---- Flash attention: 1 block = (b,h, 64 query rows); 4 waves x 16 rows ---
// Key blocks of 32: S = two 16x16 tiles (4 dense WMMA), shared softmax stats,
// P is a full 16x32 A-fragment -> P*V is 4 dense WMMA.  8 WMMA / 32 keys.
__global__ __launch_bounds__(128) void flash_attn_kernel(
    const bf16_t* __restrict__ q, const bf16_t* __restrict__ k,
    const bf16_t* __restrict__ vT, bf16_t* __restrict__ out) {
  const int lane = threadIdx.x & 31;
  const int wave = threadIdx.x >> 5;
  const int hsel = lane >> 4;
  const int l16  = lane & 15;
  const int bh = blockIdx.y;                 // b*HEADS + h
  const int bb = bh / HEADS, hh = bh % HEADS;
  const int qrow = blockIdx.x * 64 + wave * 16;

  const bf16_t* qp = q  + ((size_t)bh * SEQ + qrow) * HD;
  const bf16_t* kp = k  + (size_t)bh * SEQ * HD;
  const bf16_t* vp = vT + (size_t)bh * HD * SEQ;

  // Q fragments for hd = 0..31 and 32..63 (reused across all key blocks)
  v16bf qf0 = load_a_frag(qp, HD, 0, lane);
  v16bf qf1 = load_a_frag(qp, HD, 32, lane);

  v8f zf = {};
  v8f o[4];
#pragma unroll
  for (int t = 0; t < 4; ++t) o[t] = zf;
  float mrun[8], lrun[8];
#pragma unroll
  for (int r = 0; r < 8; ++r) { mrun[r] = -1e30f; lrun[r] = 0.0f; }

  __shared__ bf16_t ptile[4][16][32];  // per-wave P transpose staging (16x32)

  for (int kb = 0; kb < SEQ / 32; ++kb) {
    // --- S = Q * K^T for 32 keys: two 16-key column tiles ------------------
    const bf16_t* kblk0 = kp + (size_t)(kb * 32) * HD;       // keys 0..15
    const bf16_t* kblk1 = kp + (size_t)(kb * 32 + 16) * HD;  // keys 16..31
    v8f s0 = zf, s1 = zf;
    s0 = wmma_bf16(qf0, load_b_frag(kblk0, HD, 0, lane), s0);
    s0 = wmma_bf16(qf1, load_b_frag(kblk0, HD, 32, lane), s0);
    s1 = wmma_bf16(qf0, load_b_frag(kblk1, HD, 0, lane), s1);
    s1 = wmma_bf16(qf1, load_b_frag(kblk1, HD, 32, lane), s1);

    // --- online softmax over 32 keys; row r+8*hsel lives in one 16-lane half
    float p0[8], p1[8], csc[8];
#pragma unroll
    for (int r = 0; r < 8; ++r) {
      float sv0 = s0[r] * 0.125f;  // hd^-0.5
      float sv1 = s1[r] * 0.125f;
      float mx = fmaxf(sv0, sv1);  // combine tiles before cross-lane reduce
      mx = fmaxf(mx, __shfl_xor(mx, 1, 32));
      mx = fmaxf(mx, __shfl_xor(mx, 2, 32));
      mx = fmaxf(mx, __shfl_xor(mx, 4, 32));
      mx = fmaxf(mx, __shfl_xor(mx, 8, 32));
      float mnew = fmaxf(mrun[r], mx);
      csc[r] = __expf(mrun[r] - mnew);
      p0[r]  = __expf(sv0 - mnew);
      p1[r]  = __expf(sv1 - mnew);
      float rs = p0[r] + p1[r];
      rs += __shfl_xor(rs, 1, 32);
      rs += __shfl_xor(rs, 2, 32);
      rs += __shfl_xor(rs, 4, 32);
      rs += __shfl_xor(rs, 8, 32);
      lrun[r] = lrun[r] * csc[r] + rs;
      mrun[r] = mnew;
    }
#pragma unroll
    for (int t = 0; t < 4; ++t)
#pragma unroll
      for (int r = 0; r < 8; ++r) o[t][r] *= csc[r];

    // --- P: C-layout -> full 16x32 A-layout via per-wave LDS tile ----------
#pragma unroll
    for (int r = 0; r < 8; ++r) {
      ptile[wave][r + 8 * hsel][l16]      = (bf16_t)p0[r];
      ptile[wave][r + 8 * hsel][16 + l16] = (bf16_t)p1[r];
    }
    v16bf pf = load_a_frag(&ptile[wave][0][0], 32, 0, lane);

    // --- O += P * V  (V^T [hd][n]: 16 contiguous keys per lane, no EXEC div)
#pragma unroll
    for (int t = 0; t < 4; ++t) {
      v16bf vf = load_b_frag(vp + (size_t)(t * 16) * SEQ, SEQ, kb * 32, lane);
      o[t] = wmma_bf16(pf, vf, o[t]);
    }
  }

  float inv[8];
#pragma unroll
  for (int r = 0; r < 8; ++r) inv[r] = 1.0f / lrun[r];

#pragma unroll
  for (int t = 0; t < 4; ++t)
#pragma unroll
    for (int r = 0; r < 8; ++r) {
      int tokrow = bb * SEQ + qrow + r + 8 * hsel;
      int col    = hh * HD + t * 16 + l16;
      out[(size_t)tokrow * DIM + col] = (bf16_t)(o[t][r] * inv[r]);
    }
}

// ---------------------------------------------------------------------------
extern "C" void kernel_launch(void* const* d_in, const int* in_sizes, int n_in,
                              void* d_out, int out_size, void* d_ws, size_t ws_size,
                              hipStream_t stream) {
  const float* x      = (const float*)d_in[0];
  const float* ln1_g  = (const float*)d_in[1];
  const float* ln1_b  = (const float*)d_in[2];
  const float* w_qkv  = (const float*)d_in[3];
  const float* b_qkv  = (const float*)d_in[4];
  const float* w_proj = (const float*)d_in[5];
  const float* b_proj = (const float*)d_in[6];
  const float* gamma1 = (const float*)d_in[7];
  const float* ln2_g  = (const float*)d_in[8];
  const float* ln2_b  = (const float*)d_in[9];
  const float* w_fc1  = (const float*)d_in[10];
  const float* b_fc1  = (const float*)d_in[11];
  const float* w_fc2  = (const float*)d_in[12];
  const float* b_fc2  = (const float*)d_in[13];
  const float* gamma2 = (const float*)d_in[14];

  char* ws = (char*)d_ws;
  size_t off = 0;
  auto take = [&](size_t bytes) -> void* {
    void* p = ws + off;
    off = (off + bytes + 255) & ~(size_t)255;
    return p;
  };

  bf16_t* wqkvT  = (bf16_t*)take((size_t)3 * DIM * DIM * 2);     // [2304][768]
  bf16_t* wprojT = (bf16_t*)take((size_t)DIM * DIM * 2);         // [768][768]
  bf16_t* wfc1T  = (bf16_t*)take((size_t)HIDDEN * DIM * 2);      // [3072][768]
  bf16_t* wfc2T  = (bf16_t*)take((size_t)DIM * HIDDEN * 2);      // [768][3072]
  bf16_t* lnbuf  = (bf16_t*)take((size_t)TOKENS * DIM * 2);      // LN1 then LN2
  bf16_t* region1 = (bf16_t*)take((size_t)TOKENS * HIDDEN * 2);  // q,k,vT -> h
  bf16_t* region2 = (bf16_t*)take((size_t)TOKENS * 3 * DIM * 2); // qkv -> attn
  float*  x1      = (float*)take((size_t)TOKENS * DIM * 4);      // post-attn residual

  bf16_t* qbuf  = region1;
  bf16_t* kbuf  = region1 + (size_t)TOKENS * DIM;
  bf16_t* vTbuf = region1 + (size_t)2 * TOKENS * DIM;
  bf16_t* hbuf  = region1;          // reuse after q/k/vT consumed
  bf16_t* qkvbuf  = region2;
  bf16_t* attnbuf = region2;        // reuse after qkv consumed

  // 1) weight transpose + cast
  {
    size_t n;
    n = (size_t)DIM * 3 * DIM;
    transpose_cast_kernel<<<dim3((unsigned)((n + 255) / 256)), 256, 0, stream>>>(w_qkv, wqkvT, DIM, 3 * DIM);
    n = (size_t)DIM * DIM;
    transpose_cast_kernel<<<dim3((unsigned)((n + 255) / 256)), 256, 0, stream>>>(w_proj, wprojT, DIM, DIM);
    n = (size_t)DIM * HIDDEN;
    transpose_cast_kernel<<<dim3((unsigned)((n + 255) / 256)), 256, 0, stream>>>(w_fc1, wfc1T, DIM, HIDDEN);
    n = (size_t)HIDDEN * DIM;
    transpose_cast_kernel<<<dim3((unsigned)((n + 255) / 256)), 256, 0, stream>>>(w_fc2, wfc2T, HIDDEN, DIM);
  }

  // 2) LN1
  ln_kernel<<<TOKENS, 256, 0, stream>>>(x, ln1_g, ln1_b, lnbuf);

  // 3) QKV GEMM (+bias) -> bf16
  gemm_bf16_kernel<0><<<dim3((3 * DIM) / 128, TOKENS / 128), 256, 0, stream>>>(
      lnbuf, wqkvT, b_qkv, nullptr, nullptr, qkvbuf, nullptr, TOKENS, 3 * DIM, DIM);

  // 4) split/repack into per-head q, k, v^T
  repack_qkv_kernel<<<(TOKENS * DIM + 255) / 256, 256, 0, stream>>>(qkvbuf, qbuf, kbuf, vTbuf);

  // 5) flash attention
  flash_attn_kernel<<<dim3(SEQ / 64, BATCH * HEADS), 128, 0, stream>>>(qbuf, kbuf, vTbuf, attnbuf);

  // 6) proj GEMM + bias + gamma1 * (...) + residual(x) -> x1 (f32)
  gemm_bf16_kernel<1><<<dim3(DIM / 128, TOKENS / 128), 256, 0, stream>>>(
      attnbuf, wprojT, b_proj, x, gamma1, nullptr, x1, TOKENS, DIM, DIM);

  // 7) LN2
  ln_kernel<<<TOKENS, 256, 0, stream>>>(x1, ln2_g, ln2_b, lnbuf);

  // 8) FC1 GEMM + bias + exact GELU -> bf16
  gemm_bf16_kernel<2><<<dim3(HIDDEN / 128, TOKENS / 128), 256, 0, stream>>>(
      lnbuf, wfc1T, b_fc1, nullptr, nullptr, hbuf, nullptr, TOKENS, HIDDEN, DIM);

  // 9) FC2 GEMM + bias + gamma2 * (...) + residual(x1) -> d_out (f32)
  gemm_bf16_kernel<1><<<dim3(DIM / 128, TOKENS / 128), 256, 0, stream>>>(
      hbuf, wfc2T, b_fc2, x1, gamma2, nullptr, (float*)d_out, TOKENS, DIM, HIDDEN);
}
